// EIRNN_1262720385523
// MI455X (gfx1250) — matrix operand order repaired and checked by reading, not compiled, not run on hardware
//
#include <hip/hip_runtime.h>
#include <hip/hip_bf16.h>

// ---------------------------------------------------------------------------
// EI-RNN scan for MI455X (gfx1250), wave32 + WMMA f16 (f32 accumulate).
//   HIDDEN=1024, E_SIZE=819, INPUT=128, SEQ=512, BATCH=128
//   total K per step = 1024 (recurrent) + 128 (input) = 1152
// ---------------------------------------------------------------------------

#define HIDDEN   1024
#define E_SIZE   819
#define INPUT    128
#define SEQ      512
#define BATCH    128
#define KTOT     1152            // HIDDEN + INPUT
#define KTILES   (KTOT / 32)     // 36
#define NTILES   (HIDDEN / 16)   // 64
#define LDSTRIDE 1160            // 1152 + 8 halves pad (breaks bank alignment)

#define ALPHA_F     0.2f
#define ONE_MALPHA  0.8f
#define SIGMA_EFF_F 0.03162277660168379f   // sqrt(2*0.2)*0.05

typedef _Float16 v16h __attribute__((ext_vector_type(16)));
typedef _Float16 v8h  __attribute__((ext_vector_type(8)));
typedef float    v8f  __attribute__((ext_vector_type(8)));

// ---------------------------------------------------------------------------
// Prep: build combined effective weight in WMMA B-fragment order (f16).
//   Wt[k][h] = k < 1024 ? |w_rec[h][k]| * dale_sign(k) * (k!=h)
//                       : |w_in[h][k-1024]|
// Stored as tiles WB[kt][nt][lane][0..15] so each lane's 16 halves of a
// 32x16 B fragment are one contiguous 32-byte load.
//   lane <16 : K = kt*32    + j , N = nt*16 + lane
//   lane>=16 : K = kt*32+16 + j , N = nt*16 + lane-16
// ---------------------------------------------------------------------------
__global__ __launch_bounds__(256) void eirnn_prep(const float* __restrict__ w_in,
                                                  const float* __restrict__ w_rec,
                                                  _Float16* __restrict__ wb) {
    int gid = blockIdx.x * 256 + threadIdx.x;
    if (gid >= KTILES * NTILES * 32) return;
    int lane  = gid & 31;
    int nt    = (gid >> 5) & 63;
    int kt    = gid >> 11;
    int n     = nt * 16 + (lane & 15);
    int kbase = kt * 32 + (lane >> 4) * 16;

    v16h v;
#pragma unroll
    for (int j = 0; j < 16; ++j) {
        int   k = kbase + j;
        float val;
        if (k < HIDDEN) {
            float s = (k < E_SIZE) ? 1.0f : -1.0f;
            val = __builtin_fabsf(w_rec[(size_t)n * HIDDEN + k]) * s;
            if (k == n) val = 0.0f;
        } else {
            val = __builtin_fabsf(w_in[(size_t)n * INPUT + (k - HIDDEN)]);
        }
        v[j] = (_Float16)val;
    }
    *(v16h*)(wb + (size_t)gid * 16) = v;
}

// ---------------------------------------------------------------------------
// Persistent scan kernel: one block = one 16-row batch tile, 8 waves.
// Wave w owns output columns [w*128, w*128+128) as 8 WMMA N-tiles.
// A-matrix [16 x 1152] (f16) lives in LDS: [out_{t-1} | x_t].
// State kept in f32 registers (8 N-tiles x 8 rows per lane).
// ---------------------------------------------------------------------------
__global__ __launch_bounds__(256) void eirnn_scan(const float* __restrict__ x,
                                                  const float* __restrict__ noise,
                                                  const float* __restrict__ b_in,
                                                  const float* __restrict__ b_rec,
                                                  const _Float16* __restrict__ wb,
                                                  float* __restrict__ out) {
    __shared__ _Float16 smem[16 * LDSTRIDE];

    const int tid   = threadIdx.x;
    const int wave  = tid >> 5;
    const int lane  = tid & 31;
    const int col   = lane & 15;      // N within tile / A row index
    const int rhalf = lane >> 4;      // 0: rows 0-7 & K 0-15 ; 1: rows 8-15 & K 16-31
    const int brow0 = blockIdx.x * 16;

    // zero A buffer (out_0 = 0)
    for (int i = tid; i < 16 * LDSTRIDE; i += 256) smem[i] = (_Float16)0.0f;
    __syncthreads();

    // per-lane bias for the 8 owned columns (constant over time)
    float bias[8];
#pragma unroll
    for (int q = 0; q < 8; ++q) {
        int h = wave * 128 + q * 16 + col;
        bias[q] = b_in[h] + b_rec[h];
    }

    // f32 state registers: st[ntile][row]
    float st[8][8];
#pragma unroll
    for (int q = 0; q < 8; ++q)
#pragma unroll
        for (int r = 0; r < 8; ++r) st[q][r] = 0.0f;

    const int xrow = tid >> 4;          // 0..15
    const int xi0  = (tid & 15) * 8;    // 0..120

    for (int t = 0; t < SEQ; ++t) {
        // ---- stage x_t (f32 -> f16) into A[:, 1024:1152] ----
        {
            const float* xp = x + ((size_t)t * BATCH + brow0 + xrow) * INPUT + xi0;
            v8h hv;
#pragma unroll
            for (int j = 0; j < 8; ++j) hv[j] = (_Float16)xp[j];
            *(v8h*)&smem[xrow * LDSTRIDE + HIDDEN + xi0] = hv;
        }
        __syncthreads();   // out_{t-1} + x_t visible to all waves

        // ---- GEMM: D[16 x 128(wave)] = A[16 x 1152] * B[1152 x 128] ----
        v8f acc[8];
#pragma unroll
        for (int q = 0; q < 8; ++q) acc[q] = (v8f)(0.0f);

        const int akoff = rhalf * 16;
        for (int kt = 0; kt < KTILES; ++kt) {
            const _Float16* ap = &smem[col * LDSTRIDE + kt * 32 + akoff];
            v8h alo = *(const v8h*)(ap);
            v8h ahi = *(const v8h*)(ap + 8);
            v16h afrag = __builtin_shufflevector(alo, ahi,
                0, 1, 2, 3, 4, 5, 6, 7, 8, 9, 10, 11, 12, 13, 14, 15);
#pragma unroll
            for (int q = 0; q < 8; ++q) {
                const int ntg = wave * 8 + q;
                v16h bfrag = *(const v16h*)(wb + ((size_t)(kt * 64 + ntg) * 32 + lane) * 16);
                acc[q] = __builtin_amdgcn_wmma_f32_16x16x32_f16(
                    false, afrag, false, bfrag, (short)0, acc[q], false, false);
            }
        }
        __syncthreads();   // all A reads done before epilogue overwrites A

        // ---- epilogue: state update, ReLU, write out (global + LDS f16) ----
#pragma unroll
        for (int q = 0; q < 8; ++q) {
            const int h = wave * 128 + q * 16 + col;
#pragma unroll
            for (int r = 0; r < 8; ++r) {
                const int    bl = rhalf * 8 + r;  // local batch row (= D row M)
                const size_t gi = ((size_t)t * BATCH + brow0 + bl) * HIDDEN + h;
                float total = acc[q][r] + bias[q];
                float s = st[q][r] * ONE_MALPHA + total * ALPHA_F
                          + SIGMA_EFF_F * noise[gi];
                st[q][r] = s;
                float o = s > 0.0f ? s : 0.0f;
                out[gi] = o;
                smem[bl * LDSTRIDE + h] = (_Float16)o;
            }
        }
        // next iteration's leading __syncthreads (after x staging) fences these writes
    }
}

// ---------------------------------------------------------------------------
extern "C" void kernel_launch(void* const* d_in, const int* in_sizes, int n_in,
                              void* d_out, int out_size, void* d_ws, size_t ws_size,
                              hipStream_t stream) {
    const float* x     = (const float*)d_in[0];
    const float* noise = (const float*)d_in[1];
    const float* w_in  = (const float*)d_in[2];
    const float* b_in  = (const float*)d_in[3];
    const float* w_rec = (const float*)d_in[4];
    const float* b_rec = (const float*)d_in[5];
    float*       out   = (float*)d_out;
    _Float16*    wb    = (_Float16*)d_ws;   // 1152*1024 f16 = 2.36 MB (L2-resident)

    int prep_threads = KTILES * NTILES * 32;                 // 73728
    eirnn_prep<<<(prep_threads + 255) / 256, 256, 0, stream>>>(w_in, w_rec, wb);
    eirnn_scan<<<BATCH / 16, 256, 0, stream>>>(x, noise, b_in, b_rec, wb, out);
}
